// AttentionLayer_66056597012665
// MI455X (gfx1250) — matrix operand order repaired and checked by reading, not compiled, not run on hardware
//
#include <hip/hip_runtime.h>

// ---------------------------------------------------------------------------
// Fused multi-head attention for MI455X (gfx1250, wave32, WMMA).
//   B=4, N=4096 points; Kn=32 neighbors; F=64; H=16 heads; D=16.
// Roofline: 155 MB traffic (6.7us @ 23.3TB/s) vs ~35 GFLOP f16 WMMA -> fuse
// projections + attention so inp is read exactly once; use
// v_wmma_f32_16x16x32_f16 for the 32x64 @ 64x16 per-head projections.
// Note: the K-projection bias is softmax-shift-invariant (Q.bk is constant
// over the 32 neighbors), so it is dropped entirely.
// ---------------------------------------------------------------------------

typedef __attribute__((ext_vector_type(16))) _Float16 v16h;
typedef __attribute__((ext_vector_type(8)))  float    v8f;

#define HEADS 16
#define KN    32
#define FF    64
#define NPTS  (4 * 4096)

// swizzled halves per weight matrix: [head][kchunk][lane][16 halves]
#define WSWZ_ELEMS (HEADS * 2 * 32 * 16)   // 16384 halves = 32KB

// per-wave f32 scratch: K tile 32x20 (16B-aligned rows) + Q row + weights
#define KROW   20
#define WSTRIDE 704                        // 640 + 16 + 32 + pad

// ---------------------------------------------------------------------------
// Kernel 1: swizzle W[64][256] (f32) into CDNA5 WMMA B-matrix layout (f16).
// B (32x16 per k-chunk): lane L holds column N = L%16; half t holds
// K-row = 32*c + 8*(L/16) + (t&7) + 16*(t/8).   dst index = ((h*2+c)*32 + L)*16 + t.
// ---------------------------------------------------------------------------
__global__ __launch_bounds__(256) void wswz_kernel(
    const float* __restrict__ Wk, const float* __restrict__ Wv,
    const float* __restrict__ Wq, _Float16* __restrict__ dst) {
  int idx = blockIdx.x * 256 + threadIdx.x;          // 0 .. 3*16384-1
  int m = idx / WSWZ_ELEMS;                          // 0=K, 1=V, 2=Q
  int r = idx % WSWZ_ELEMS;
  int t = r & 15;
  int L = (r >> 4) & 31;
  int c = (r >> 9) & 1;
  int h = (r >> 10) & 15;
  const float* W = (m == 0) ? Wk : (m == 1 ? Wv : Wq);
  float v = W[(32 * c + 8 * (L >> 4) + (t & 7) + ((t >> 3) << 4)) * 256 + h * 16 + (L & 15)];
  dst[idx] = (_Float16)v;
}

// ---------------------------------------------------------------------------
// Helpers
// ---------------------------------------------------------------------------

// Build one f16 A-matrix tile (16x32) register fragment from row-major f32.
// Lane L = row M (mod 16); halves: K = 8*(L/16) + 32*c + {0..7, 16..23}.
__device__ __forceinline__ v16h load_a16(const float* __restrict__ s) {
  float4 f0 = *(const float4*)(s);
  float4 f1 = *(const float4*)(s + 4);
  float4 f2 = *(const float4*)(s + 16);
  float4 f3 = *(const float4*)(s + 20);
  v16h a;
  a[0] = (_Float16)f0.x;  a[1] = (_Float16)f0.y;
  a[2] = (_Float16)f0.z;  a[3] = (_Float16)f0.w;
  a[4] = (_Float16)f1.x;  a[5] = (_Float16)f1.y;
  a[6] = (_Float16)f1.z;  a[7] = (_Float16)f1.w;
  a[8] = (_Float16)f2.x;  a[9] = (_Float16)f2.y;
  a[10] = (_Float16)f2.z; a[11] = (_Float16)f2.w;
  a[12] = (_Float16)f3.x; a[13] = (_Float16)f3.y;
  a[14] = (_Float16)f3.z; a[15] = (_Float16)f3.w;
  return a;
}

// 32B (16 halves) contiguous per-lane read of a pre-swizzled B fragment.
__device__ __forceinline__ v16h lds_b16(const _Float16* __restrict__ p) {
  union { uint4 u[2]; v16h v; } t;
  t.u[0] = *(const uint4*)(p);
  t.u[1] = *(const uint4*)(p + 8);
  return t.v;
}

#define WMMA_F16(a, b, c)                                                    \
  __builtin_amdgcn_wmma_f32_16x16x32_f16(false, (a), false, (b), (short)0,   \
                                         (c), false, false)

// ---------------------------------------------------------------------------
// Kernel 2: fused projections + attention.
// grid = 512 blocks x 256 threads (8 waves); each wave owns 4 points.
// ---------------------------------------------------------------------------
__global__ __launch_bounds__(256, 2) void attn_kernel(
    const float* __restrict__ inp, const float* __restrict__ query,
    const float* __restrict__ bq, const float* __restrict__ bv,
    const _Float16* __restrict__ wswz, float* __restrict__ out) {
  extern __shared__ __align__(16) char smem_raw[];
  _Float16* wS   = (_Float16*)smem_raw;                  // 3*16384 halves
  float* scratch = (float*)(smem_raw + 3 * WSWZ_ELEMS * 2);

  const int tid  = threadIdx.x;
  const int lane = tid & 31;
  const int wid  = tid >> 5;
  const int g    = lane >> 4;    // half-wave group
  const int ln   = lane & 15;

  // Stage all swizzled weights (96KB) into LDS, coalesced uint4 copies.
  {
    const uint4* src = (const uint4*)wswz;
    uint4* dst = (uint4*)wS;
    const int nvec = (3 * WSWZ_ELEMS * 2) / 16;          // 6144
    for (int i = tid; i < nvec; i += 256) dst[i] = src[i];
  }
  __syncthreads();

  float* KS  = scratch + wid * WSTRIDE;    // 32 rows x 20 floats
  float* QS  = KS + 32 * KROW;             // 16 floats (16B aligned)
  float* WSm = KS + 32 * KROW + 16;        // 32 floats (16B aligned)

  for (int i = 0; i < 4; ++i) {
    const int p = blockIdx.x * 32 + wid * 4 + i;         // 512*32 = 16384 pts
    const float* X  = inp + (size_t)p * (KN * FF);
    const float* qv = query + (size_t)p * FF;
    float* op = out + (size_t)p * (HEADS * 16);

    if (p + 1 < NPTS)  // pull next point's 8KB tile toward L2
      __builtin_prefetch(inp + (size_t)(p + 1) * (KN * FF) + lane * 64, 0, 1);

    // A fragments for inp tile: [row-tile t][k-chunk c]; reused by all heads.
    v16h A[2][2];
    #pragma unroll
    for (int t = 0; t < 2; ++t)
      #pragma unroll
      for (int c = 0; c < 2; ++c)
        A[t][c] = load_a16(X + (ln + 16 * t) * FF + 32 * c + 8 * g);

    // Query broadcast A fragments: only row M=0 (lanes 0,16) carries data.
    v16h QA[2];
    #pragma unroll
    for (int c = 0; c < 2; ++c) {
      v16h qa;
      #pragma unroll
      for (int e = 0; e < 16; ++e) qa[e] = (_Float16)0.0f;
      if (ln == 0) qa = load_a16(qv + 32 * c + 8 * g);
      QA[c] = qa;
    }

    #pragma unroll 1
    for (int h = 0; h < HEADS; ++h) {
      const v8f zacc = {0.f, 0.f, 0.f, 0.f, 0.f, 0.f, 0.f, 0.f};
      const int hb = (h * 2) * 32 * 16;                 // head base (halves)
      const int cs = 32 * 16;                           // chunk stride

      // Hoisted, non-divergent bias loads (lanes 16..31 read duplicates).
      const float bqv = bq[h * 16 + ln];
      const float bvv = bv[h * 16 + ln];

      // K_h = X @ Wk_h : two 16x16 C tiles, K-dim 64 as two x32 chunks.
      v8f Kc0, Kc1;
      {
        v16h b0 = lds_b16(wS + hb + lane * 16);
        v16h b1 = lds_b16(wS + hb + cs + lane * 16);
        Kc0 = WMMA_F16(A[0][1], b1, WMMA_F16(A[0][0], b0, zacc));
        Kc1 = WMMA_F16(A[1][1], b1, WMMA_F16(A[1][0], b0, zacc));
      }
      // V_h = X @ Wv_h
      v8f Vc0, Vc1;
      {
        v16h b0 = lds_b16(wS + WSWZ_ELEMS + hb + lane * 16);
        v16h b1 = lds_b16(wS + WSWZ_ELEMS + hb + cs + lane * 16);
        Vc0 = WMMA_F16(A[0][1], b1, WMMA_F16(A[0][0], b0, zacc));
        Vc1 = WMMA_F16(A[1][1], b1, WMMA_F16(A[1][0], b0, zacc));
      }
      // Q_h = q @ Wq_h (row 0 of C tile is the valid row)
      v8f Qc;
      {
        v16h b0 = lds_b16(wS + 2 * WSWZ_ELEMS + hb + lane * 16);
        v16h b1 = lds_b16(wS + 2 * WSWZ_ELEMS + hb + cs + lane * 16);
        Qc = WMMA_F16(QA[1], b1, WMMA_F16(QA[0], b0, zacc));
      }

      // Spill K (C layout: elem j -> row j+8g, col ln) to padded LDS.
      // bk is omitted: it shifts every score equally -> softmax invariant.
      #pragma unroll
      for (int j = 0; j < 8; ++j) {
        KS[(j + 8 * g) * KROW + ln]      = Kc0[j];
        KS[(j + 8 * g + 16) * KROW + ln] = Kc1[j];
      }
      // Q row with bias; fold 1/sqrt(D)=0.25 into Q.
      if (lane < 16) QS[lane] = 0.25f * (Qc[0] + bqv);
      __builtin_amdgcn_wave_barrier();   // LDS pipe is in-order per wave

      // scores: lane k = lane, s_k = sum_d Q[d] * K[k][d]  (float4 LDS reads)
      const float4 q0 = *(const float4*)(QS + 0);
      const float4 q1 = *(const float4*)(QS + 4);
      const float4 q2 = *(const float4*)(QS + 8);
      const float4 q3 = *(const float4*)(QS + 12);
      const float4 k0 = *(const float4*)(KS + lane * KROW + 0);
      const float4 k1 = *(const float4*)(KS + lane * KROW + 4);
      const float4 k2 = *(const float4*)(KS + lane * KROW + 8);
      const float4 k3 = *(const float4*)(KS + lane * KROW + 12);
      float s = q0.x * k0.x + q0.y * k0.y + q0.z * k0.z + q0.w * k0.w +
                q1.x * k1.x + q1.y * k1.y + q1.z * k1.z + q1.w * k1.w +
                q2.x * k2.x + q2.y * k2.y + q2.z * k2.z + q2.w * k2.w +
                q3.x * k3.x + q3.y * k3.y + q3.z * k3.z + q3.w * k3.w;

      // softmax over all 32 lanes
      float m = s;
      #pragma unroll
      for (int off = 16; off > 0; off >>= 1)
        m = fmaxf(m, __shfl_xor(m, off));
      float e = __expf(s - m);
      float tot = e;
      #pragma unroll
      for (int off = 16; off > 0; off >>= 1) tot += __shfl_xor(tot, off);
      WSm[lane] = e / tot;
      __builtin_amdgcn_wave_barrier();

      // out_d = sum_k w_k * V[k][d]; V read straight from C-layout regs.
      // Softmax weights sum to 1, so the V bias folds in additively.
      const float4 w0 = *(const float4*)(WSm + 8 * g);        // rows 8g..+3
      const float4 w1 = *(const float4*)(WSm + 8 * g + 4);    // rows +4..7
      const float4 w2 = *(const float4*)(WSm + 8 * g + 16);   // rows +16..19
      const float4 w3 = *(const float4*)(WSm + 8 * g + 20);   // rows +20..23
      float acc = w0.x * Vc0[0] + w0.y * Vc0[1] + w0.z * Vc0[2] +
                  w0.w * Vc0[3] + w1.x * Vc0[4] + w1.y * Vc0[5] +
                  w1.z * Vc0[6] + w1.w * Vc0[7] +
                  w2.x * Vc1[0] + w2.y * Vc1[1] + w2.z * Vc1[2] +
                  w2.w * Vc1[3] + w3.x * Vc1[4] + w3.y * Vc1[5] +
                  w3.z * Vc1[6] + w3.w * Vc1[7];
      acc += __shfl_xor(acc, 16);        // combine the two half-wave groups
      if (lane < 16) op[h * 16 + lane] = acc + bvv;
      __builtin_amdgcn_wave_barrier();
    }
  }
}

// ---------------------------------------------------------------------------
extern "C" void kernel_launch(void* const* d_in, const int* in_sizes, int n_in,
                              void* d_out, int out_size, void* d_ws,
                              size_t ws_size, hipStream_t stream) {
  const float* inp   = (const float*)d_in[0];
  const float* query = (const float*)d_in[1];
  const float* Wq    = (const float*)d_in[2];
  const float* bq    = (const float*)d_in[3];
  const float* Wk    = (const float*)d_in[4];
  const float* bv    = (const float*)d_in[7];
  const float* Wv    = (const float*)d_in[6];
  float* out = (float*)d_out;
  _Float16* wswz = (_Float16*)d_ws;   // 96KB of scratch used

  // 1) swizzle weights into WMMA B layout (3*16384 halves)
  wswz_kernel<<<(3 * WSWZ_ELEMS) / 256, 256, 0, stream>>>(Wk, Wv, Wq, wswz);

  // 2) fused attention; dynamic LDS: 96KB weights + 8 waves * 2816B scratch
  const size_t smem = (size_t)(3 * WSWZ_ELEMS * 2) +
                      8 * WSTRIDE * sizeof(float);       // 120832 B
  (void)hipFuncSetAttribute((const void*)attn_kernel,
                            hipFuncAttributeMaxDynamicSharedMemorySize,
                            (int)smem);
  attn_kernel<<<NPTS / 32, 256, smem, stream>>>(inp, query, bq, bv, wswz,
                                                out);
}
